// Random_SpatialMix_51101520888172
// MI455X (gfx1250) — matrix operand order repaired and checked by reading, not compiled, not run on hardware
//
#include <hip/hip_runtime.h>
#include <hip/hip_bf16.h>

#define DEVINL static __device__ __forceinline__

typedef __attribute__((ext_vector_type(16))) __bf16 v16bf;
typedef __attribute__((ext_vector_type(8)))  __bf16 v8bf;
typedef __attribute__((ext_vector_type(8)))  float  v8f;
typedef __attribute__((ext_vector_type(8)))  unsigned short v8u16;

constexpr int B_ = 8, T_ = 1024, C_ = 768, H_ = 32, W_ = 32;
constexpr int BT = B_ * T_;                 // GEMM M = 8192
constexpr float INV_T = 1.0f / (float)T_;

// wkv chunking
constexpr int NC = 16;                      // chunks along T
constexpr int CL = T_ / NC;                 // 64 steps per chunk

// GEMM block tiling
constexpr int BM = 128, BN = 64, BK = 32;
constexpr int LDSW = 40;                    // halves per LDS row (32 + 8 pad -> conflict-free)

DEVINL unsigned short f32_to_bf16_bits(float f) {
  unsigned int u = __float_as_uint(f);
  unsigned int r = 0x7FFFu + ((u >> 16) & 1u);   // round-to-nearest-even
  return (unsigned short)((u + r) >> 16);
}

DEVINL v16bf concat16(v8u16 lo, v8u16 hi) {
  v8bf a = __builtin_bit_cast(v8bf, lo);
  v8bf b = __builtin_bit_cast(v8bf, hi);
  return __builtin_shufflevector(a, b, 0,1,2,3,4,5,6,7,8,9,10,11,12,13,14,15);
}

// CDNA5 async global->LDS copy, 16B per lane.
// Pass the real LDS pointer: ptrtoint captures the shared array (so the
// "memory" clobber is honored against it), and for LDS generic addresses
// the low 32 bits ARE the LDS byte address (ISA 10.2 aperture rules).
DEVINL void async_copy_b128(unsigned short* lptr, const void* gptr) {
  asm volatile("global_load_async_to_lds_b128 %0, %1, off"
               :: "v"((unsigned)(uintptr_t)lptr),
                  "v"((unsigned long long)(uintptr_t)gptr)
               : "memory");
}
DEVINL void wait_async_all() {
  asm volatile("s_wait_asynccnt 0x0" ::: "memory");
}

// ---- LDS fragment loads (row stride LDSW halves) ----
// A fragment 16x32 (MxK): lanes 0-15 row=l: halves = K{0..7,16..23}; lanes 16-31: K{8..15,24..31}
DEVINL v16bf lds_a_frag(const unsigned short* s, int baseH, int mr0, int l, int hi) {
  const unsigned short* p = s + baseH + (mr0 + l) * LDSW + hi * 8;
  v8u16 lo = *(const v8u16*)(p);
  v8u16 h2 = *(const v8u16*)(p + 16);
  return concat16(lo, h2);
}
// B fragment 32x16 (KxN), stored as W rows (K contiguous): lanes 0-15 K{0..7,16..23}; lanes 16-31 K{8..15,24..31}
DEVINL v16bf lds_b_frag(const unsigned short* s, int baseH, int nr0, int l, int hi) {
  const unsigned short* p = s + baseH + (nr0 + l) * LDSW + hi * 8;
  v8u16 lo = *(const v8u16*)(p);
  v8u16 h2 = *(const v8u16*)(p + 16);
  return concat16(lo, h2);
}

// ---- async staging: global (row stride C_) -> LDS tile (row stride LDSW) ----
// A tile: BM rows x BK halves = 128x4 16B-chunks = 512 chunks -> 2 per thread
DEVINL void stage_a(unsigned short* smem, const unsigned short* A,
                    int m0, int k0, int baseH, int tid) {
#pragma unroll
  for (int j = 0; j < 2; ++j) {
    int chunk = tid * 2 + j;
    int row = chunk >> 2, kc = chunk & 3;
    const unsigned short* g = A + (size_t)(m0 + row) * C_ + k0 + kc * 8;
    async_copy_b128(smem + baseH + row * LDSW + kc * 8, g);
  }
}
// B tile: BN rows x BK halves = 64x4 chunks = 256 -> 1 per thread
DEVINL void stage_b(unsigned short* smem, const unsigned short* Wt,
                    int n0, int k0, int baseH, int tid) {
  int row = tid >> 2, kc = tid & 3;
  const unsigned short* g = Wt + (size_t)(n0 + row) * C_ + k0 + kc * 8;
  async_copy_b128(smem + baseH + row * LDSW + kc * 8, g);
}

DEVINL v8f wmma_bf16(v16bf a, v16bf b, v8f c) {
  return __builtin_amdgcn_wmma_f32_16x16x32_bf16(false, a, false, b, (short)0, c, false, false);
}

// ---------------- elementwise prep kernels ----------------

__global__ __launch_bounds__(256) void cvt_bf16_kernel(const float* __restrict__ src,
                                                       unsigned short* __restrict__ dst, int n) {
  int i = blockIdx.x * blockDim.x + threadIdx.x;
  if (i < n) dst[i] = f32_to_bf16_bits(src[i]);
}

__global__ __launch_bounds__(256) void shift_mix_kernel(
    const float* __restrict__ x,
    const float* __restrict__ mk, const float* __restrict__ mv, const float* __restrict__ mr,
    unsigned short* __restrict__ xk, unsigned short* __restrict__ xv,
    unsigned short* __restrict__ xr) {
  int idx = blockIdx.x * blockDim.x + threadIdx.x;
  if (idx >= B_ * T_ * C_) return;
  int c = idx % C_;
  int t = (idx / C_) % T_;
  int b = idx / (C_ * T_);
  int h = t >> 5, w = t & (W_ - 1);
  const int g = C_ / 4;                       // 192; 4*g == C_ (no passthrough group)
  int hs = h, ws = w; bool ok;
  if      (c <     g) { ws = w - 1; ok = (ws >= 0); }
  else if (c < 2 * g) { ws = w + 1; ok = (ws < W_); }
  else if (c < 3 * g) { hs = h - 1; ok = (hs >= 0); }
  else                { hs = h + 1; ok = (hs < H_); }
  float xx = 0.0f;
  if (ok) xx = x[((size_t)b * T_ + (size_t)(hs * W_ + ws)) * C_ + c];
  float xc = x[idx];
  float k = mk[c], v = mv[c], r = mr[c];
  xk[idx] = f32_to_bf16_bits(xc * k + xx * (1.0f - k));
  xv[idx] = f32_to_bf16_bits(xc * v + xx * (1.0f - v));
  xr[idx] = f32_to_bf16_bits(xc * r + xx * (1.0f - r));
}

// Per-channel wkv params with permutation folded in:
// output channel j uses decay/first[rev[j]]; since j = perm[i] <=> rev[j] = i:
__global__ void wu_prep_kernel(const float* __restrict__ decay, const float* __restrict__ first,
                               const int* __restrict__ perm,
                               float* __restrict__ wch, float* __restrict__ uch) {
  int i = blockIdx.x * blockDim.x + threadIdx.x;
  if (i < C_) {
    int j = perm[i];
    wch[j] = -__expf(decay[i] * INV_T);
    uch[j] = first[i] * INV_T;
  }
}

// ---------------- fused k/v/r tiled WMMA GEMM ----------------
// grid (C_/BN, BT/BM), block 256 = 8 waves (4 M-groups x 2 N-groups), wave tile 32x32.

__global__ __launch_bounds__(256) void gemm_kvr_kernel(
    const unsigned short* __restrict__ xk, const unsigned short* __restrict__ xv,
    const unsigned short* __restrict__ xr,
    const unsigned short* __restrict__ wk, const unsigned short* __restrict__ wv,
    const unsigned short* __restrict__ wr,
    float* __restrict__ ko, float* __restrict__ vo, float* __restrict__ so) {
  __shared__ unsigned short smem[3 * BM * LDSW + 3 * BN * LDSW];   // 46080 B
  const int A0 = 0, A1 = BM * LDSW, A2 = 2 * BM * LDSW;
  const int Bb = 3 * BM * LDSW;
  const int B0 = Bb, B1 = Bb + BN * LDSW, B2 = Bb + 2 * BN * LDSW;

  const int tid  = threadIdx.x;
  const int lane = tid & 31, wave = tid >> 5;
  const int mg = wave & 3, ng = wave >> 2;
  const int mr0 = mg * 32, nr0 = ng * 32;
  const int l = lane & 15, hi = lane >> 4;
  const int m0 = blockIdx.y * BM, n0 = blockIdx.x * BN;

  // Make the shared array visibly written/address-taken, then order vs async writes.
  ((volatile unsigned short*)smem)[tid] = 0;
  __syncthreads();

  const v8f vzero = {0,0,0,0,0,0,0,0};
  v8f acc[3][2][2];
#pragma unroll
  for (int m = 0; m < 3; ++m)
#pragma unroll
    for (int i = 0; i < 2; ++i)
#pragma unroll
      for (int j = 0; j < 2; ++j) acc[m][i][j] = vzero;

  const unsigned short* Amats[3] = {xk, xv, xr};
  const unsigned short* Wmats[3] = {wk, wv, wr};
  const int Abase[3] = {A0, A1, A2};
  const int Bbase[3] = {B0, B1, B2};

  for (int k0 = 0; k0 < C_; k0 += BK) {
#pragma unroll
    for (int m = 0; m < 3; ++m) {
      stage_a(smem, Amats[m], m0, k0, Abase[m], tid);
      stage_b(smem, Wmats[m], n0, k0, Bbase[m], tid);
    }
    wait_async_all();
    __syncthreads();
#pragma unroll
    for (int m = 0; m < 3; ++m) {
      v16bf a0 = lds_a_frag(smem, Abase[m], mr0,      l, hi);
      v16bf a1 = lds_a_frag(smem, Abase[m], mr0 + 16, l, hi);
      v16bf b0 = lds_b_frag(smem, Bbase[m], nr0,      l, hi);
      v16bf b1 = lds_b_frag(smem, Bbase[m], nr0 + 16, l, hi);
      acc[m][0][0] = wmma_bf16(a0, b0, acc[m][0][0]);
      acc[m][0][1] = wmma_bf16(a0, b1, acc[m][0][1]);
      acc[m][1][0] = wmma_bf16(a1, b0, acc[m][1][0]);
      acc[m][1][1] = wmma_bf16(a1, b1, acc[m][1][1]);
    }
    __syncthreads();   // all reads done before next stage overwrites LDS
  }

  // C/D layout: VGPR r -> row +r+8*hi, col +l
#pragma unroll
  for (int mi = 0; mi < 2; ++mi)
#pragma unroll
    for (int ni = 0; ni < 2; ++ni) {
      int rbase = m0 + mr0 + mi * 16 + hi * 8;
      int cbase = n0 + nr0 + ni * 16 + l;
#pragma unroll
      for (int r = 0; r < 8; ++r) {
        size_t o = (size_t)(rbase + r) * C_ + (size_t)cbase;
        ko[o] = acc[0][mi][ni][r];
        vo[o] = acc[1][mi][ni][r];
        so[o] = 1.0f / (1.0f + __expf(-acc[2][mi][ni][r]));
      }
    }
}

// ---------------- chunked wkv scan ----------------
// Unscaled recurrence A' = e^w A + e^k v, B' = e^w B + e^k with constant e^w.
// Chunk transition: A_end = e^{w*CL} A_0 + S_chunk, S_chunk = scan from zero state,
// carried in log-scaled (a,b,p) form: (A,B) = (a,b)*e^p.

__global__ __launch_bounds__(256) void wkv_pass1_kernel(
    const float* __restrict__ kk, const float* __restrict__ vv,
    const float* __restrict__ wch,
    float* __restrict__ cs_a, float* __restrict__ cs_b, float* __restrict__ cs_p) {
  int tid = blockIdx.x * blockDim.x + threadIdx.x;     // (b, chunk, c)
  if (tid >= B_ * NC * C_) return;
  int c = tid % C_;
  int chunk = (tid / C_) % NC;
  int b = tid / (C_ * NC);
  float w = wch[c];
  float a = 0.0f, bb = 0.0f, p = -1e38f;
  size_t base = ((size_t)b * T_ + (size_t)chunk * CL) * C_ + (size_t)c;
  for (int t = 0; t < CL; ++t) {
    size_t o = base + (size_t)t * C_;
    float kt = kk[o], vt = vv[o];
    float q2 = fmaxf(p + w, kt);
    float e1 = __expf(p + w - q2);
    float e2 = __expf(kt - q2);
    a = e1 * a + e2 * vt;
    bb = e1 * bb + e2;
    p = q2;
  }
  cs_a[tid] = a; cs_b[tid] = bb; cs_p[tid] = p;
}

__global__ __launch_bounds__(256) void wkv_combine_kernel(
    const float* __restrict__ cs_a, const float* __restrict__ cs_b, const float* __restrict__ cs_p,
    const float* __restrict__ wch,
    float* __restrict__ ci_a, float* __restrict__ ci_b, float* __restrict__ ci_p) {
  int tid = blockIdx.x * blockDim.x + threadIdx.x;     // (b, c)
  if (tid >= B_ * C_) return;
  int c = tid % C_;
  int b = tid / C_;
  float wL = wch[c] * (float)CL;
  float a = 0.0f, bb = 0.0f, p = -1e38f;
  for (int i = 0; i < NC; ++i) {
    size_t o = ((size_t)b * NC + i) * C_ + (size_t)c;
    ci_a[o] = a; ci_b[o] = bb; ci_p[o] = p;            // state entering chunk i
    float pa = p + wL;
    float lp = cs_p[o];
    float q  = fmaxf(pa, lp);
    float e1 = __expf(pa - q);
    float e2 = __expf(lp - q);
    a  = e1 * a  + e2 * cs_a[o];
    bb = e1 * bb + e2 * cs_b[o];
    p  = q;
  }
}

__global__ __launch_bounds__(256) void wkv_pass2_kernel(
    const float* __restrict__ kk, const float* __restrict__ vv, const float* __restrict__ sr,
    const float* __restrict__ wch, const float* __restrict__ uch,
    const float* __restrict__ ci_a, const float* __restrict__ ci_b, const float* __restrict__ ci_p,
    unsigned short* __restrict__ z) {
  int tid = blockIdx.x * blockDim.x + threadIdx.x;     // (b, chunk, c)
  if (tid >= B_ * NC * C_) return;
  int c = tid % C_;
  int chunk = (tid / C_) % NC;
  int b = tid / (C_ * NC);
  float w = wch[c], u = uch[c];
  float a = ci_a[tid], bb = ci_b[tid], p = ci_p[tid];
  size_t base = ((size_t)b * T_ + (size_t)chunk * CL) * C_ + (size_t)c;
  for (int t = 0; t < CL; ++t) {
    size_t o = base + (size_t)t * C_;
    float kt = kk[o], vt = vv[o];
    float q  = fmaxf(p, u + kt);
    float e1 = __expf(p - q);
    float e2 = __expf(u + kt - q);
    float y  = (e1 * a + e2 * vt) / (e1 * bb + e2);
    float q2 = fmaxf(p + w, kt);
    e1 = __expf(p + w - q2);
    e2 = __expf(kt - q2);
    a  = e1 * a + e2 * vt;
    bb = e1 * bb + e2;
    p  = q2;
    z[o] = f32_to_bf16_bits(sr[o] * y);
  }
}

// ---------------- output tiled WMMA GEMM: out = z @ Wo^T ----------------

__global__ __launch_bounds__(256) void gemm_out_kernel(
    const unsigned short* __restrict__ z, const unsigned short* __restrict__ wo,
    float* __restrict__ out) {
  __shared__ unsigned short smem[BM * LDSW + BN * LDSW];   // 15360 B
  const int A0 = 0, B0 = BM * LDSW;

  const int tid  = threadIdx.x;
  const int lane = tid & 31, wave = tid >> 5;
  const int mg = wave & 3, ng = wave >> 2;
  const int mr0 = mg * 32, nr0 = ng * 32;
  const int l = lane & 15, hi = lane >> 4;
  const int m0 = blockIdx.y * BM, n0 = blockIdx.x * BN;

  ((volatile unsigned short*)smem)[tid] = 0;
  __syncthreads();

  const v8f vzero = {0,0,0,0,0,0,0,0};
  v8f acc[2][2] = {{vzero, vzero}, {vzero, vzero}};

  for (int k0 = 0; k0 < C_; k0 += BK) {
    stage_a(smem, z,  m0, k0, A0, tid);
    stage_b(smem, wo, n0, k0, B0, tid);
    wait_async_all();
    __syncthreads();
    v16bf a0 = lds_a_frag(smem, A0, mr0,      l, hi);
    v16bf a1 = lds_a_frag(smem, A0, mr0 + 16, l, hi);
    v16bf b0 = lds_b_frag(smem, B0, nr0,      l, hi);
    v16bf b1 = lds_b_frag(smem, B0, nr0 + 16, l, hi);
    acc[0][0] = wmma_bf16(a0, b0, acc[0][0]);
    acc[0][1] = wmma_bf16(a0, b1, acc[0][1]);
    acc[1][0] = wmma_bf16(a1, b0, acc[1][0]);
    acc[1][1] = wmma_bf16(a1, b1, acc[1][1]);
    __syncthreads();
  }
#pragma unroll
  for (int mi = 0; mi < 2; ++mi)
#pragma unroll
    for (int ni = 0; ni < 2; ++ni) {
      int rbase = m0 + mr0 + mi * 16 + hi * 8;
      int cbase = n0 + nr0 + ni * 16 + l;
#pragma unroll
      for (int r = 0; r < 8; ++r)
        out[(size_t)(rbase + r) * C_ + (size_t)cbase] = acc[mi][ni][r];
    }
}

// ---------------- host launcher ----------------

extern "C" void kernel_launch(void* const* d_in, const int* in_sizes, int n_in,
                              void* d_out, int out_size, void* d_ws, size_t ws_size,
                              hipStream_t stream) {
  (void)in_sizes; (void)n_in; (void)out_size; (void)ws_size;
  const float* x     = (const float*)d_in[0];
  const float* w_key = (const float*)d_in[1];
  const float* w_val = (const float*)d_in[2];
  const float* w_rec = (const float*)d_in[3];
  const float* w_out = (const float*)d_in[4];
  const float* decay = (const float*)d_in[5];
  const float* first = (const float*)d_in[6];
  const float* mk    = (const float*)d_in[7];
  const float* mv    = (const float*)d_in[8];
  const float* mr    = (const float*)d_in[9];
  const int*   perm  = (const int*)d_in[10];

  char* ws = (char*)d_ws;
  size_t off = 0;
  auto take = [&](size_t bytes) -> char* {
    char* p = ws + off;
    off += (bytes + 255) & ~(size_t)255;
    return p;
  };
  const size_t wBytes = (size_t)C_ * C_ * 2;
  const size_t xBytes = (size_t)BT * C_ * 2;
  const size_t fBytes = (size_t)BT * C_ * 4;
  const size_t sBytes = (size_t)B_ * NC * C_ * 4;
  unsigned short* wkb = (unsigned short*)take(wBytes);
  unsigned short* wvb = (unsigned short*)take(wBytes);
  unsigned short* wrb = (unsigned short*)take(wBytes);
  unsigned short* wob = (unsigned short*)take(wBytes);
  unsigned short* xkb = (unsigned short*)take(xBytes);   // reused as z after gemm_kvr
  unsigned short* xvb = (unsigned short*)take(xBytes);
  unsigned short* xrb = (unsigned short*)take(xBytes);
  float* kf = (float*)take(fBytes);
  float* vf = (float*)take(fBytes);
  float* sf = (float*)take(fBytes);
  float* cs_a = (float*)take(sBytes);
  float* cs_b = (float*)take(sBytes);
  float* cs_p = (float*)take(sBytes);
  float* ci_a = (float*)take(sBytes);
  float* ci_b = (float*)take(sBytes);
  float* ci_p = (float*)take(sBytes);
  float* wch = (float*)take((size_t)C_ * 4);
  float* uch = (float*)take((size_t)C_ * 4);

  const int nW = C_ * C_;
  cvt_bf16_kernel<<<(nW + 255) / 256, 256, 0, stream>>>(w_key, wkb, nW);
  cvt_bf16_kernel<<<(nW + 255) / 256, 256, 0, stream>>>(w_val, wvb, nW);
  cvt_bf16_kernel<<<(nW + 255) / 256, 256, 0, stream>>>(w_rec, wrb, nW);
  cvt_bf16_kernel<<<(nW + 255) / 256, 256, 0, stream>>>(w_out, wob, nW);

  const int nX = B_ * T_ * C_;
  shift_mix_kernel<<<(nX + 255) / 256, 256, 0, stream>>>(x, mk, mv, mr, xkb, xvb, xrb);
  wu_prep_kernel<<<(C_ + 255) / 256, 256, 0, stream>>>(decay, first, perm, wch, uch);

  dim3 gGemm(C_ / BN, BT / BM);   // 12 x 64 blocks
  gemm_kvr_kernel<<<gGemm, 256, 0, stream>>>(xkb, xvb, xrb, wkb, wvb, wrb, kf, vf, sf);

  const int nChunk = B_ * NC * C_;
  wkv_pass1_kernel<<<(nChunk + 255) / 256, 256, 0, stream>>>(kf, vf, wch, cs_a, cs_b, cs_p);
  wkv_combine_kernel<<<(B_ * C_ + 255) / 256, 256, 0, stream>>>(cs_a, cs_b, cs_p, wch,
                                                                ci_a, ci_b, ci_p);
  wkv_pass2_kernel<<<(nChunk + 255) / 256, 256, 0, stream>>>(kf, vf, sf, wch, uch,
                                                             ci_a, ci_b, ci_p,
                                                             xkb /* z reuses xk region */);

  gemm_out_kernel<<<gGemm, 256, 0, stream>>>(xkb, wob, (float*)d_out);
}
